// multihead_attention_54150947668558
// MI455X (gfx1250) — compile-verified, matrix operand order checked by reading, not run on hardware
//
#include <hip/hip_runtime.h>
#include <hip/hip_bf16.h>

// Problem constants (match reference)
#define B_   4
#define LQ_  1024
#define LK_  1024
#define DQ_  64
#define DK_  512
#define DM_  1024
#define H_   16
#define DH_  64

typedef __attribute__((ext_vector_type(16))) __bf16 v16bf;
typedef __attribute__((ext_vector_type(8)))  float  v8f;

union FragAB {
    v16bf v;
    uint4 q[2];
    unsigned short u[16];
    __bf16 h[16];
};
union FragC {
    v8f v;
    float f[8];
};

__device__ __forceinline__ unsigned short f2bf(float f) {
    __bf16 h = (__bf16)f;                 // native cvt (v_cvt_pk_bf16_f32 path)
    union { __bf16 h; unsigned short u; } x; x.h = h;
    return x.u;
}
__device__ __forceinline__ float bf2f(unsigned short b) {
    union { __bf16 h; unsigned short u; } x; x.u = b;
    return (float)x.h;
}
__device__ __forceinline__ float softplus_f(float x) {
    return (x > 20.f) ? x : log1pf(__expf(x));
}

#define WMMA_BF16(A_, B_frag, C_) \
    __builtin_amdgcn_wmma_f32_16x16x32_bf16(false, (A_).v, false, (B_frag).v, (short)0, (C_).v, false, false)

// ---------------------------------------------------------------------------
// Elementwise f32 -> bf16 convert (for activations y, x: row-major kept)
// ---------------------------------------------------------------------------
__global__ void cvt_bf16(const float* __restrict__ in, unsigned short* __restrict__ out, int n) {
    int i = blockIdx.x * 256 + threadIdx.x;
    if (i < n) out[i] = f2bf(in[i]);
}

// ---------------------------------------------------------------------------
// Weight convert + transpose: in f32 [K,N] row-major -> out bf16 [N,K].
// Output-linear indexing (coalesced writes; strided reads hit L2 once).
// ---------------------------------------------------------------------------
__global__ void cvt_transpose_bf16(const float* __restrict__ in, unsigned short* __restrict__ out,
                                   int K, int N) {
    int i = blockIdx.x * 256 + threadIdx.x;
    if (i < K * N) {
        int n = i / K, k = i % K;
        out[i] = f2bf(in[(size_t)k * N + n]);
    }
}

// ---------------------------------------------------------------------------
// C[M,N] = act(A[M,K] @ W[K,N] + bias[N]); A bf16 [M,K] row-major,
// Wt bf16 [N,K] (pre-transposed), C bf16 [M,N].
// Block = 128 threads = 4 waves arranged 2x2; each wave owns a 64x64
// macro-tile: 4x4 WMMA 16x16x32 tiles -> 16 WMMAs per 16 b128 loads/K-step.
// No LDS: both fragment types are contiguous 2x16B loads per lane.
// grid: (N/128, M/128)
// ---------------------------------------------------------------------------
__global__ __launch_bounds__(128)
void gemm_bias_act_bf16(const unsigned short* __restrict__ A,
                        const unsigned short* __restrict__ Wt,
                        const float* __restrict__ bias,
                        unsigned short* __restrict__ C,
                        int M, int N, int K, int act)
{
    const int tid  = threadIdx.x;
    const int wave = tid >> 5;
    const int lane = tid & 31;
    const int half = lane >> 4;   // 0: lanes 0-15, 1: lanes 16-31
    const int lr   = lane & 15;
    const int wm   = wave >> 1, wn = wave & 1;
    const int m0   = blockIdx.y * 128 + wm * 64;
    const int n0   = blockIdx.x * 128 + wn * 64;

    FragC acc[4][4];
#pragma unroll
    for (int mi = 0; mi < 4; ++mi)
#pragma unroll
        for (int ni = 0; ni < 4; ++ni)
#pragma unroll
            for (int i = 0; i < 8; ++i) acc[mi][ni].f[i] = 0.f;

    const unsigned short* ap[4];
    const unsigned short* bp[4];
#pragma unroll
    for (int t = 0; t < 4; ++t) {
        ap[t] = A  + (size_t)(m0 + t * 16 + lr) * K + half * 8;
        bp[t] = Wt + (size_t)(n0 + t * 16 + lr) * K + half * 8;
    }

    for (int k0 = 0; k0 < K; k0 += 32) {
        FragAB a[4], b[4];
#pragma unroll
        for (int t = 0; t < 4; ++t) {
            a[t].q[0] = *(const uint4*)(ap[t] + k0);
            a[t].q[1] = *(const uint4*)(ap[t] + k0 + 16);
        }
#pragma unroll
        for (int t = 0; t < 4; ++t) {
            b[t].q[0] = *(const uint4*)(bp[t] + k0);
            b[t].q[1] = *(const uint4*)(bp[t] + k0 + 16);
        }
#pragma unroll
        for (int mi = 0; mi < 4; ++mi)
#pragma unroll
            for (int ni = 0; ni < 4; ++ni)
                acc[mi][ni].v = WMMA_BF16(a[mi], b[ni], acc[mi][ni]);
    }

#pragma unroll
    for (int mi = 0; mi < 4; ++mi) {
#pragma unroll
        for (int ni = 0; ni < 4; ++ni) {
            const int   col = n0 + ni * 16 + lr;
            const float bv  = bias[col];
#pragma unroll
            for (int i = 0; i < 8; ++i) {
                int   row = m0 + mi * 16 + half * 8 + i;
                float r   = acc[mi][ni].f[i] + bv;
                if (act) r = softplus_f(r);
                C[(size_t)row * N + col] = f2bf(r);
            }
        }
    }
}

// ---------------------------------------------------------------------------
// scores[b,h,q,k] = (qh.kh)/sqrt(DK) + (1-mask)*-1e9  (f32 into attn slot).
// Both fragments contiguous (B's column n = a key-row of k). 2x2 blocking.
// grid: (LK/64, LQ/64, B*H), block 128.
// ---------------------------------------------------------------------------
__global__ __launch_bounds__(128)
void attn_scores(const unsigned short* __restrict__ q,
                 const unsigned short* __restrict__ k,
                 const int* __restrict__ mask,
                 float* __restrict__ attn)
{
    const int tid  = threadIdx.x;
    const int wave = tid >> 5;
    const int lane = tid & 31;
    const int half = lane >> 4;
    const int lr   = lane & 15;
    const int wm   = wave >> 1, wn = wave & 1;
    const int bh   = blockIdx.z;
    const int b    = bh / H_, h = bh % H_;
    const int qt0  = blockIdx.y * 64 + wm * 32;
    const int kt0  = blockIdx.x * 64 + wn * 32;

    FragC acc00, acc01, acc10, acc11;
#pragma unroll
    for (int i = 0; i < 8; ++i) {
        acc00.f[i] = 0.f; acc01.f[i] = 0.f; acc10.f[i] = 0.f; acc11.f[i] = 0.f;
    }

    const unsigned short* q0p = q + (size_t)(b * LQ_ + qt0 + lr) * DM_ + h * DH_ + half * 8;
    const unsigned short* q1p = q0p + (size_t)16 * DM_;
    const unsigned short* k0p = k + (size_t)(b * LK_ + kt0 + lr) * DM_ + h * DH_ + half * 8;
    const unsigned short* k1p = k0p + (size_t)16 * DM_;

#pragma unroll
    for (int d0 = 0; d0 < DH_; d0 += 32) {
        FragAB a0, a1, b0, b1;
        a0.q[0] = *(const uint4*)(q0p + d0);  a0.q[1] = *(const uint4*)(q0p + d0 + 16);
        a1.q[0] = *(const uint4*)(q1p + d0);  a1.q[1] = *(const uint4*)(q1p + d0 + 16);
        b0.q[0] = *(const uint4*)(k0p + d0);  b0.q[1] = *(const uint4*)(k0p + d0 + 16);
        b1.q[0] = *(const uint4*)(k1p + d0);  b1.q[1] = *(const uint4*)(k1p + d0 + 16);
        acc00.v = WMMA_BF16(a0, b0, acc00);
        acc01.v = WMMA_BF16(a0, b1, acc01);
        acc10.v = WMMA_BF16(a1, b0, acc10);
        acc11.v = WMMA_BF16(a1, b1, acc11);
    }

    const float rscale = 0.044194173824159216f;  // 1/sqrt(512)
    FragC* accs[2][2] = { { &acc00, &acc01 }, { &acc10, &acc11 } };
#pragma unroll
    for (int mi = 0; mi < 2; ++mi) {
#pragma unroll
        for (int ni = 0; ni < 2; ++ni) {
            const int kcol = kt0 + ni * 16 + lr;
            FragC* ac = accs[mi][ni];
#pragma unroll
            for (int i = 0; i < 8; ++i) {
                int   qrow = qt0 + mi * 16 + half * 8 + i;
                float s    = ac->f[i] * rscale;
                int   mv   = mask[(size_t)(b * LQ_ + qrow) * LK_ + kcol];
                s += (1.0f - (float)mv) * -1e9f;
                attn[((size_t)bh * LQ_ + qrow) * LK_ + kcol] = s;
            }
        }
    }
}

// ---------------------------------------------------------------------------
// In-place softmax over last dim (LK_=1024). One block (256 thr) per row.
// ---------------------------------------------------------------------------
__global__ __launch_bounds__(256)
void softmax_rows(float* __restrict__ attn)
{
    __shared__ float red[256];
    float* p = attn + (size_t)blockIdx.x * LK_;
    const int tid = threadIdx.x;

    float vals[4];
    float m = -1e30f;
#pragma unroll
    for (int i = 0; i < 4; ++i) { vals[i] = p[tid + i * 256]; m = fmaxf(m, vals[i]); }
    red[tid] = m; __syncthreads();
    for (int s = 128; s > 0; s >>= 1) {
        if (tid < s) red[tid] = fmaxf(red[tid], red[tid + s]);
        __syncthreads();
    }
    m = red[0]; __syncthreads();

    float sum = 0.f;
#pragma unroll
    for (int i = 0; i < 4; ++i) { vals[i] = __expf(vals[i] - m); sum += vals[i]; }
    red[tid] = sum; __syncthreads();
    for (int s = 128; s > 0; s >>= 1) {
        if (tid < s) red[tid] += red[tid + s];
        __syncthreads();
    }
    float inv = 1.0f / red[0];
#pragma unroll
    for (int i = 0; i < 4; ++i) p[tid + i * 256] = vals[i] * inv;
}

// ---------------------------------------------------------------------------
// out[b,h,q,d] = (attn @ v_h)[q,d] + LK*y[b,q,d].
// A = attn f32 (converted to bf16 on the fly; this GEMM is bound by the
// 268MB attn re-read anyway). B = vtr bf16 [B,H,DH,LK] (pre-transposed),
// so B-fragments are contiguous. Wave: 16q x 32d (1x2 tiles).
// Block 128 = 4 waves 2x2 -> 32q x 64d per block. grid: (1, LQ/32, B*H)
// ---------------------------------------------------------------------------
__global__ __launch_bounds__(128)
void attn_ctx(const float* __restrict__ attn,
              const unsigned short* __restrict__ vtr,
              const float* __restrict__ y,
              float* __restrict__ out)
{
    const int tid  = threadIdx.x;
    const int wave = tid >> 5;
    const int lane = tid & 31;
    const int half = lane >> 4;
    const int lr   = lane & 15;
    const int wm   = wave >> 1, wn = wave & 1;
    const int bh   = blockIdx.z;
    const int b    = bh / H_;
    const int q0   = blockIdx.y * 32 + wm * 16;
    const int d0   = wn * 32;

    FragC acc0, acc1;
#pragma unroll
    for (int i = 0; i < 8; ++i) { acc0.f[i] = 0.f; acc1.f[i] = 0.f; }

    const float* arow = attn + ((size_t)bh * LQ_ + q0 + lr) * LK_ + half * 8;
    const unsigned short* v0p = vtr + ((size_t)bh * DH_ + d0 + lr) * LK_ + half * 8;
    const unsigned short* v1p = v0p + (size_t)16 * LK_;

#pragma unroll 2
    for (int k0 = 0; k0 < LK_; k0 += 32) {
        FragAB a, b0, b1;
        const float* ap = arow + k0;
#pragma unroll
        for (int j = 0; j < 8; ++j) a.h[j]     = (__bf16)ap[j];
#pragma unroll
        for (int j = 0; j < 8; ++j) a.h[8 + j] = (__bf16)ap[16 + j];
        b0.q[0] = *(const uint4*)(v0p + k0);  b0.q[1] = *(const uint4*)(v0p + k0 + 16);
        b1.q[0] = *(const uint4*)(v1p + k0);  b1.q[1] = *(const uint4*)(v1p + k0 + 16);
        acc0.v = WMMA_BF16(a, b0, acc0);
        acc1.v = WMMA_BF16(a, b1, acc1);
    }

    FragC* accs[2] = { &acc0, &acc1 };
#pragma unroll
    for (int ni = 0; ni < 2; ++ni) {
        const int dcol = d0 + ni * 16 + lr;
        FragC* ac = accs[ni];
#pragma unroll
        for (int i = 0; i < 8; ++i) {
            int   qrow = q0 + half * 8 + i;
            float r = ac->f[i] + (float)LK_ * y[(size_t)(b * LQ_ + qrow) * DQ_ + dcol];
            out[((size_t)bh * LQ_ + qrow) * DH_ + dcol] = r;
        }
    }
}

// ---------------------------------------------------------------------------
// From v bf16 [B*LK, DM]: write vr f32 [B,H,LK,DH] (output 3) and
// vtr bf16 [B,H,DH,LK] (feeds attn_ctx B-fragments).
// ---------------------------------------------------------------------------
__global__ void v_outputs(const unsigned short* __restrict__ v,
                          float* __restrict__ vr,
                          unsigned short* __restrict__ vtr)
{
    int i = blockIdx.x * 256 + threadIdx.x;  // over B*H*LK*DH
    // vr[i] with i = (((b*H+h)*LK + kk)*DH + d)
    {
        int d = i & 63; int t = i >> 6; int kk = t & (LK_ - 1); int bh = t >> 10;
        int h = bh & (H_ - 1), b = bh >> 4;
        vr[i] = bf2f(v[(size_t)(b * LK_ + kk) * DM_ + h * DH_ + d]);
    }
    // vtr[i] with i = (((b*H+h)*DH + d)*LK + kk)
    {
        int kk = i & (LK_ - 1); int t = i >> 10; int d = t & 63; int bh = t >> 6;
        int h = bh & (H_ - 1), b = bh >> 4;
        vtr[i] = v[(size_t)(b * LK_ + kk) * DM_ + h * DH_ + d];
    }
}

// ---------------------------------------------------------------------------
extern "C" void kernel_launch(void* const* d_in, const int* in_sizes, int n_in,
                              void* d_out, int out_size, void* d_ws, size_t ws_size,
                              hipStream_t stream) {
    (void)in_sizes; (void)n_in; (void)out_size; (void)ws_size;

    const float* y    = (const float*)d_in[0];
    const float* x    = (const float*)d_in[1];
    const int*   mask = (const int*)d_in[2];
    const float* W[9]  = { (const float*)d_in[3],  (const float*)d_in[5],  (const float*)d_in[7],
                           (const float*)d_in[9],  (const float*)d_in[11], (const float*)d_in[13],
                           (const float*)d_in[15], (const float*)d_in[17], (const float*)d_in[19] };
    const float* bias[9] = { (const float*)d_in[4],  (const float*)d_in[6],  (const float*)d_in[8],
                             (const float*)d_in[10], (const float*)d_in[12], (const float*)d_in[14],
                             (const float*)d_in[16], (const float*)d_in[18], (const float*)d_in[20] };
    const int Wk[9] = { DQ_, DM_, DM_, DK_, DM_, DM_, DK_, DM_, DM_ };

    // d_out partitions: out | attn | vr
    float* out_p  = (float*)d_out;
    float* attn_p = out_p + (size_t)B_*H_*LQ_*DH_;
    float* vr_p   = attn_p + (size_t)B_*H_*LQ_*LK_;

    // workspace carve-up (bf16 buffers)
    char* ws = (char*)d_ws;
    size_t off = 0;
    auto alloc = [&](size_t bytes) -> void* {
        void* p = ws + off;
        off = (off + bytes + 255) & ~(size_t)255;
        return p;
    };
    unsigned short* yb  = (unsigned short*)alloc((size_t)B_*LQ_*DQ_*2);
    unsigned short* xb  = (unsigned short*)alloc((size_t)B_*LK_*DK_*2);
    unsigned short* Wb[9];
    for (int i = 0; i < 9; ++i) Wb[i] = (unsigned short*)alloc((size_t)Wk[i]*DM_*2);
    unsigned short* h1  = (unsigned short*)alloc((size_t)B_*LQ_*DM_*2);
    unsigned short* h2  = (unsigned short*)alloc((size_t)B_*LQ_*DM_*2);
    unsigned short* qb  = (unsigned short*)alloc((size_t)B_*LQ_*DM_*2);
    unsigned short* kb  = (unsigned short*)alloc((size_t)B_*LK_*DM_*2);
    unsigned short* vb  = (unsigned short*)alloc((size_t)B_*LK_*DM_*2);
    unsigned short* vtr = (unsigned short*)alloc((size_t)B_*H_*DH_*LK_*2);

    // --- f32 -> bf16 conversions; weights transposed to [N,K] ---
    {
        int n = B_*LQ_*DQ_;
        cvt_bf16<<<dim3((n + 255) / 256), dim3(256), 0, stream>>>(y, yb, n);
        n = B_*LK_*DK_;
        cvt_bf16<<<dim3((n + 255) / 256), dim3(256), 0, stream>>>(x, xb, n);
    }
    for (int i = 0; i < 9; ++i) {
        int n = Wk[i] * DM_;
        cvt_transpose_bf16<<<dim3((n + 255) / 256), dim3(256), 0, stream>>>(W[i], Wb[i], Wk[i], DM_);
    }

    const int M = B_*LQ_;  // 4096 rows for all MLPs
    auto gemm = [&](const unsigned short* A, const unsigned short* Wt, const float* bi,
                    unsigned short* C, int K, int act) {
        gemm_bias_act_bf16<<<dim3(DM_/128, M/128), dim3(128), 0, stream>>>(A, Wt, bi, C, M, DM_, K, act);
    };

    // q = MLP(y)
    gemm(yb, Wb[0], bias[0], h1, DQ_, 1);
    gemm(h1, Wb[1], bias[1], h2, DM_, 1);
    gemm(h2, Wb[2], bias[2], qb, DM_, 0);
    // k = MLP(x)
    gemm(xb, Wb[3], bias[3], h1, DK_, 1);
    gemm(h1, Wb[4], bias[4], h2, DM_, 1);
    gemm(h2, Wb[5], bias[5], kb, DM_, 0);
    // v = MLP(x)
    gemm(xb, Wb[6], bias[6], h1, DK_, 1);
    gemm(h1, Wb[7], bias[7], h2, DM_, 1);
    gemm(h2, Wb[8], bias[8], vb, DM_, 0);

    // vr output + v transpose for the PV GEMM
    {
        int n = B_*H_*LK_*DH_;
        v_outputs<<<dim3(n/256), dim3(256), 0, stream>>>(vb, vr_p, vtr);
    }
    // scores -> attn (f32, pre-softmax)
    attn_scores<<<dim3(LK_/64, LQ_/64, B_*H_), dim3(128), 0, stream>>>(qb, kb, mask, attn_p);
    // softmax in place
    softmax_rows<<<dim3(B_*H_*LQ_), dim3(256), 0, stream>>>(attn_p);
    // ctx + LK*y -> out
    attn_ctx<<<dim3(1, LQ_/32, B_*H_), dim3(128), 0, stream>>>(attn_p, vtr, y, out_p);
}
